// QwenGroupedQueryAttention_44100724195922
// MI455X (gfx1250) — matrix operand order repaired and checked by reading, not compiled, not run on hardware
//
#include <hip/hip_runtime.h>
#include <hip/hip_bf16.h>

typedef __attribute__((ext_vector_type(16))) __bf16 v16bf;
typedef __attribute__((ext_vector_type(8)))  float  v8f;

union frag16 {
  v16bf v;
  uint4 u[2];
};

__device__ __forceinline__ v8f wmma_bf16(const v16bf& a, const v16bf& b, const v8f& c) {
  // D = A(16x32 bf16) * B(32x16 bf16) + C(16x16 f32)
  return __builtin_amdgcn_wmma_f32_16x16x32_bf16(false, a, false, b, (short)0, c, false, false);
}

__device__ __forceinline__ v8f v8f_zero() {
  v8f z = {0.f, 0.f, 0.f, 0.f, 0.f, 0.f, 0.f, 0.f};
  return z;
}

// gfx1250 async global->LDS copy, 16B per lane, tracked by ASYNCcnt.
// saddr form: LDS destination from lds_off; global source is base + voff_bytes
__device__ __forceinline__ void async_copy_b128(unsigned lds_off, const void* base,
                                                unsigned voff_bytes) {
  asm volatile("global_load_async_to_lds_b128 %0, %1, %2"
               :
               : "v"(lds_off), "v"(voff_bytes), "s"(base)
               : "memory");
}
#define WAIT_ASYNC(n) asm volatile("s_wait_asynccnt " #n ::: "memory")

// ---------------------------------------------------------------------------
// Elementwise fp32 -> bf16.
// ---------------------------------------------------------------------------
__launch_bounds__(256)
__global__ void to_bf16(const float* __restrict__ X, __bf16* __restrict__ Xb) {
  const size_t i = ((size_t)blockIdx.x * 256 + threadIdx.x) * 8;
  float4 f0 = ((const float4*)(X + i))[0];
  float4 f1 = ((const float4*)(X + i))[1];
  union { __bf16 h8[8]; uint4 u; } o;
  o.h8[0] = (__bf16)f0.x; o.h8[1] = (__bf16)f0.y;
  o.h8[2] = (__bf16)f0.z; o.h8[3] = (__bf16)f0.w;
  o.h8[4] = (__bf16)f1.x; o.h8[5] = (__bf16)f1.y;
  o.h8[6] = (__bf16)f1.z; o.h8[7] = (__bf16)f1.w;
  *(uint4*)(Xb + i) = o.u;
}

// ---------------------------------------------------------------------------
// W (K x N) fp32 row-major  ->  Wt (N x K) bf16 (column-major of W).
// ---------------------------------------------------------------------------
__launch_bounds__(256)
__global__ void transpose_bf16(const float* __restrict__ W, __bf16* __restrict__ Wt,
                               int K, int N) {
  __shared__ float tile[32][33];
  const int bn = blockIdx.x * 32;
  const int bk = blockIdx.y * 32;
  const int tx = threadIdx.x & 31;
  const int ty = threadIdx.x >> 5;
#pragma unroll
  for (int j = 0; j < 32; j += 8)
    tile[ty + j][tx] = W[(size_t)(bk + ty + j) * N + (bn + tx)];
  __syncthreads();
#pragma unroll
  for (int j = 0; j < 32; j += 8)
    Wt[(size_t)(bn + ty + j) * K + (bk + tx)] = (__bf16)tile[tx][ty + j];
}

// ---------------------------------------------------------------------------
// bf16 x bf16 -> fp32 GEMM via WMMA, async-LDS staging, manually unrolled
// double buffering (buffer index is compile-time constant in each half).
// A: M x K row-major bf16.  Bt: N x K row-major bf16 (B column-major).
// Block: 256 threads (8 waves). Tile 128x128, K-step 32; K/32 must be even.
// ---------------------------------------------------------------------------
__launch_bounds__(256)
__global__ void gemm_bb(const __bf16* __restrict__ A, const __bf16* __restrict__ Bt,
                        const float* __restrict__ bias, float* __restrict__ C,
                        int M, int N, int K) {
  constexpr int LDA = 40;  // padded bf16 stride: 16B-aligned frags, no bank conflicts
  __shared__ __align__(16) __bf16 As[2][128 * LDA];  // [row][k]
  __shared__ __align__(16) __bf16 Bs[2][128 * LDA];  // [col][k]

  const int tid  = threadIdx.x;
  const int wave = tid >> 5;
  const int lane = tid & 31;
  const int g    = lane >> 4;
  const int ln   = lane & 15;
  const int wm   = wave >> 1;
  const int wn   = wave & 1;
  const int m0   = blockIdx.y * 128;
  const int n0   = blockIdx.x * 128;

  // staging: 512 slots of 8 bf16 per tile; thread owns slots tid and tid+256
  const int r0  = tid >> 2;       // 0..63
  const int kp0 = (tid & 3) * 8;  // 0,8,16,24
  unsigned aL[2][2], bL[2][2];
#pragma unroll
  for (int d = 0; d < 2; ++d) {
    aL[d][0] = (unsigned)(uintptr_t)&As[d][r0 * LDA + kp0];
    aL[d][1] = (unsigned)(uintptr_t)&As[d][(r0 + 64) * LDA + kp0];
    bL[d][0] = (unsigned)(uintptr_t)&Bs[d][r0 * LDA + kp0];
    bL[d][1] = (unsigned)(uintptr_t)&Bs[d][(r0 + 64) * LDA + kp0];
  }
  const unsigned aG0 = (unsigned)(((size_t)(m0 + r0) * K + kp0) * 2);
  const unsigned aG1 = aG0 + (unsigned)((size_t)64 * K * 2);
  const unsigned bG0 = (unsigned)(((size_t)(n0 + r0) * K + kp0) * 2);
  const unsigned bG1 = bG0 + (unsigned)((size_t)64 * K * 2);

  v8f acc[2][4];
#pragma unroll
  for (int r = 0; r < 2; ++r)
#pragma unroll
    for (int f = 0; f < 4; ++f) acc[r][f] = v8f_zero();

  auto stage = [&](int d, int kchunk) {
    const unsigned ko = (unsigned)kchunk * 64u;  // 32 bf16 = 64 bytes
    async_copy_b128(aL[d][0], A, aG0 + ko);
    async_copy_b128(aL[d][1], A, aG1 + ko);
    async_copy_b128(bL[d][0], Bt, bG0 + ko);
    async_copy_b128(bL[d][1], Bt, bG1 + ko);
  };
  auto compute = [&](const __bf16* as, const __bf16* bs) {
    // A fragments: lane holds row m=ln, K = g*8+{0..7} then 16+g*8+{0..7}
    frag16 af[2];
#pragma unroll
    for (int r = 0; r < 2; ++r) {
      const __bf16* p = as + (wm * 32 + r * 16 + ln) * LDA + g * 8;
      af[r].u[0] = *(const uint4*)p;
      af[r].u[1] = *(const uint4*)(p + 16);
    }
    // B fragments: lane holds col n=ln, K = g*8+{0..7} then 16+g*8+{0..7}
    frag16 bfr[4];
#pragma unroll
    for (int f = 0; f < 4; ++f) {
      const __bf16* p = bs + (wn * 64 + f * 16 + ln) * LDA + g * 8;
      bfr[f].u[0] = *(const uint4*)p;
      bfr[f].u[1] = *(const uint4*)(p + 16);
    }
#pragma unroll
    for (int r = 0; r < 2; ++r)
#pragma unroll
      for (int f = 0; f < 4; ++f)
        acc[r][f] = wmma_bf16(af[r].v, bfr[f].v, acc[r][f]);
  };

  const int nk = K >> 5;  // even (K multiple of 64)

  stage(0, 0);
  for (int kt = 0; kt < nk; kt += 2) {
    // ---- half 1: compute buf0 (chunk kt), prefetch buf1 (chunk kt+1) ----
    stage(1, kt + 1);
    WAIT_ASYNC(4);       // oldest 4 (buf0) complete; buf1 still in flight
    __syncthreads();
    compute(&As[0][0], &Bs[0][0]);
    __syncthreads();
    // ---- half 2: compute buf1 (chunk kt+1), prefetch buf0 (chunk kt+2) ----
    if (kt + 2 < nk) {
      stage(0, kt + 2);
      WAIT_ASYNC(4);
    } else {
      WAIT_ASYNC(0);
    }
    __syncthreads();
    compute(&As[1][0], &Bs[1][0]);
    __syncthreads();
  }

  // epilogue: C-frag element i at (m = g*8+i, n = ln)
#pragma unroll
  for (int r = 0; r < 2; ++r) {
#pragma unroll
    for (int f = 0; f < 4; ++f) {
      const int col = n0 + wn * 64 + f * 16 + ln;
      const float bv = bias ? bias[col] : 0.0f;
#pragma unroll
      for (int i = 0; i < 8; ++i) {
        const int row = m0 + wm * 32 + r * 16 + g * 8 + i;
        C[(size_t)row * N + col] = acc[r][f][i] + bv;
      }
    }
  }
}

// ---------------------------------------------------------------------------
// RoPE (on K and V, matching the reference) + bf16 repack.
//   Qb: (B,16,T,128) bf16 (no rope on q per reference)
//   Kb: (B,4,T,128)  bf16, rope applied
//   Vt: (B,4,128,T)  bf16, rope applied, transposed for P*V B-fragments
// ---------------------------------------------------------------------------
__launch_bounds__(256)
__global__ void rope_pack(const float* __restrict__ qkv,
                          const float* __restrict__ fcos, const float* __restrict__ fsin,
                          __bf16* __restrict__ Qb, __bf16* __restrict__ Kb,
                          __bf16* __restrict__ Vt) {
  const int bt  = blockIdx.x;  // B*T = 4096
  const int b   = bt >> 11;
  const int t   = bt & 2047;
  const int tid = threadIdx.x;
  const float* row = qkv + (size_t)bt * 3072;

  {  // Q: 2048 floats, 8 per thread
    const int base = tid * 8;
    const int h = base >> 7;
    const int d = base & 127;
    float4 f0 = ((const float4*)(row + base))[0];
    float4 f1 = ((const float4*)(row + base))[1];
    union { __bf16 h8[8]; uint4 u; } o;
    o.h8[0] = (__bf16)f0.x; o.h8[1] = (__bf16)f0.y;
    o.h8[2] = (__bf16)f0.z; o.h8[3] = (__bf16)f0.w;
    o.h8[4] = (__bf16)f1.x; o.h8[5] = (__bf16)f1.y;
    o.h8[6] = (__bf16)f1.z; o.h8[7] = (__bf16)f1.w;
    *(uint4*)(Qb + (((size_t)(b * 16 + h) * 2048 + t) * 128 + d)) = o.u;
  }
  {  // K and V: 4 kv-heads * 64 rotation pairs, 1 pair/thread
    const int kvh = tid >> 6;
    const int i   = tid & 63;
    const float c = fcos[t * 64 + i];
    const float s = fsin[t * 64 + i];
    {  // K
      const float a  = row[2048 + kvh * 128 + 2 * i];
      const float bb = row[2048 + kvh * 128 + 2 * i + 1];
      union { __bf16 h2[2]; unsigned u; } o;
      o.h2[0] = (__bf16)(a * c - bb * s);
      o.h2[1] = (__bf16)(a * s + bb * c);
      *(unsigned*)(Kb + (((size_t)(b * 4 + kvh) * 2048 + t) * 128 + 2 * i)) = o.u;
    }
    {  // V (rope'd per reference, stored transposed)
      const float a  = row[2560 + kvh * 128 + 2 * i];
      const float bb = row[2560 + kvh * 128 + 2 * i + 1];
      const size_t ob = ((size_t)(b * 4 + kvh) * 128 + 2 * i) * 2048 + t;
      Vt[ob]        = (__bf16)(a * c - bb * s);
      Vt[ob + 2048] = (__bf16)(a * s + bb * c);
    }
  }
}

// ---------------------------------------------------------------------------
// Causal flash attention with bf16 WMMA. Grid: (T/64, H, B), 128 threads.
// Each wave owns 16 q-rows; kv in 32-wide chunks with online softmax.
// Causal masking only on the wave-uniform boundary chunks.
// ---------------------------------------------------------------------------
__launch_bounds__(128)
__global__ void attn_fwd(const __bf16* __restrict__ Qb, const __bf16* __restrict__ Kb,
                         const __bf16* __restrict__ Vt, __bf16* __restrict__ Yb) {
  const int qt   = blockIdx.x;
  const int h    = blockIdx.y;
  const int b    = blockIdx.z;
  const int wave = threadIdx.x >> 5;
  const int lane = threadIdx.x & 31;
  const int g    = lane >> 4;
  const int ln   = lane & 15;
  const int q0   = qt * 64;
  const int tq0  = q0 + wave * 16;
  const int kv   = h >> 2;  // GQA: q head h -> kv head h/4

  const __bf16* Q  = Qb + (size_t)(b * 16 + h)  * 2048 * 128;
  const __bf16* Kp = Kb + (size_t)(b * 4  + kv) * 2048 * 128;
  const __bf16* Vp = Vt + (size_t)(b * 4  + kv) * 128  * 2048;

  __shared__ __align__(16) __bf16 Ps[4][16 * 40];  // per-wave P staging

  frag16 qa[4];
#pragma unroll
  for (int kk = 0; kk < 4; ++kk) {
    const __bf16* p = Q + (size_t)(tq0 + ln) * 128 + kk * 32 + g * 8;
    qa[kk].u[0] = *(const uint4*)p;
    qa[kk].u[1] = *(const uint4*)(p + 16);
  }

  v8f O[8];
  float M[8], L[8];
#pragma unroll
  for (int c = 0; c < 8; ++c) O[c] = v8f_zero();
#pragma unroll
  for (int i = 0; i < 8; ++i) { M[i] = -1e30f; L[i] = 0.0f; }
  const float scale = 0.08838834764831845f;  // 1/sqrt(128)

  const int ktend = q0 + 64;
  for (int kt0 = 0; kt0 < ktend; kt0 += 32) {
    // S = Q * K^T for 32 kv positions
    v8f S0 = v8f_zero(), S1 = v8f_zero();
#pragma unroll
    for (int kk = 0; kk < 4; ++kk) {
      frag16 kb0, kb1;
      const __bf16* p0 = Kp + (size_t)(kt0 + ln) * 128 + kk * 32 + g * 8;
      kb0.u[0] = *(const uint4*)p0;
      kb0.u[1] = *(const uint4*)(p0 + 16);
      const __bf16* p1 = p0 + 16 * 128;
      kb1.u[0] = *(const uint4*)p1;
      kb1.u[1] = *(const uint4*)(p1 + 16);
      S0 = wmma_bf16(qa[kk].v, kb0.v, S0);
      S1 = wmma_bf16(qa[kk].v, kb1.v, S1);
    }
    float mn[8];
    if (kt0 + 31 > tq0) {  // wave-uniform: only boundary chunks need the causal mask
#pragma unroll
      for (int i = 0; i < 8; ++i) {
        const int qp = tq0 + g * 8 + i;
        float s0 = S0[i] * scale;
        float s1 = S1[i] * scale;
        if (kt0 + ln > qp)      s0 = -1e30f;
        if (kt0 + 16 + ln > qp) s1 = -1e30f;
        S0[i] = s0; S1[i] = s1;
        mn[i] = fmaxf(s0, s1);
      }
    } else {
#pragma unroll
      for (int i = 0; i < 8; ++i) {
        const float s0 = S0[i] * scale;
        const float s1 = S1[i] * scale;
        S0[i] = s0; S1[i] = s1;
        mn[i] = fmaxf(s0, s1);
      }
    }
#pragma unroll
    for (int off = 1; off < 16; off <<= 1)
#pragma unroll
      for (int i = 0; i < 8; ++i)
        mn[i] = fmaxf(mn[i], __shfl_xor(mn[i], off, 32));
    float fac[8], ps[8];
#pragma unroll
    for (int i = 0; i < 8; ++i) {
      const float Mn = fmaxf(M[i], mn[i]);
      fac[i] = __expf(M[i] - Mn);
      M[i] = Mn;
      const float p0 = __expf(S0[i] - Mn);
      const float p1 = __expf(S1[i] - Mn);
      S0[i] = p0; S1[i] = p1;
      ps[i] = p0 + p1;
    }
#pragma unroll
    for (int off = 1; off < 16; off <<= 1)
#pragma unroll
      for (int i = 0; i < 8; ++i)
        ps[i] += __shfl_xor(ps[i], off, 32);
#pragma unroll
    for (int i = 0; i < 8; ++i) L[i] = L[i] * fac[i] + ps[i];
#pragma unroll
    for (int c = 0; c < 8; ++c)
#pragma unroll
      for (int i = 0; i < 8; ++i) O[c][i] *= fac[i];

    // P (16x32) C-frag -> A-frag layout via per-wave LDS (in-order within wave)
    __bf16* pw = &Ps[wave][0];
#pragma unroll
    for (int i = 0; i < 8; ++i) {
      pw[(g * 8 + i) * 40 + ln]      = (__bf16)S0[i];
      pw[(g * 8 + i) * 40 + 16 + ln] = (__bf16)S1[i];
    }
    asm volatile("" ::: "memory");
    frag16 pa;
    const __bf16* pr = pw + ln * 40 + g * 8;
    pa.u[0] = *(const uint4*)pr;
    pa.u[1] = *(const uint4*)(pr + 16);

    // O += P * V  (V transposed: two 16B runs (kv g*8+{0..7}, 16+g*8+{0..7}) per B-fragment lane)
#pragma unroll
    for (int c = 0; c < 8; ++c) {
      frag16 vb;
      const __bf16* vp = Vp + (size_t)(c * 16 + ln) * 2048 + kt0 + g * 8;
      vb.u[0] = *(const uint4*)vp;
      vb.u[1] = *(const uint4*)(vp + 16);
      O[c] = wmma_bf16(pa.v, vb.v, O[c]);
    }
  }

  __bf16* yp = Yb + (size_t)b * 2048 * 2048;
#pragma unroll
  for (int c = 0; c < 8; ++c) {
    const int col = h * 128 + c * 16 + ln;
#pragma unroll
    for (int i = 0; i < 8; ++i) {
      const int t = tq0 + g * 8 + i;
      yp[(size_t)t * 2048 + col] = (__bf16)(O[c][i] / L[i]);
    }
  }
}

// ---------------------------------------------------------------------------
extern "C" void kernel_launch(void* const* d_in, const int* in_sizes, int n_in,
                              void* d_out, int out_size, void* d_ws, size_t ws_size,
                              hipStream_t stream) {
  (void)in_sizes; (void)n_in; (void)out_size; (void)ws_size;
  const float* x      = (const float*)d_in[0];  // (2,2048,2048)
  const float* W_attn = (const float*)d_in[1];  // (2048,3072)
  const float* b_attn = (const float*)d_in[2];  // (3072,)
  const float* W_proj = (const float*)d_in[3];  // (2048,2048)
  const float* fcos   = (const float*)d_in[4];  // (2048,64)
  const float* fsin   = (const float*)d_in[5];  // (2048,64)
  float* out = (float*)d_out;                   // (2,2048,2048)

  char* ws = (char*)d_ws;
  float*  qkv = (float*)ws;  ws += (size_t)4096 * 3072 * sizeof(float);          // 48 MB
  __bf16* Xb  = (__bf16*)ws; ws += (size_t)4096 * 2048 * sizeof(__bf16);         // 16 MB
  __bf16* Wat = (__bf16*)ws; ws += (size_t)3072 * 2048 * sizeof(__bf16);         // 12 MB
  __bf16* Wpt = (__bf16*)ws; ws += (size_t)2048 * 2048 * sizeof(__bf16);         //  8 MB
  __bf16* Qb  = (__bf16*)ws; ws += (size_t)2 * 16 * 2048 * 128 * sizeof(__bf16); // 16 MB
  __bf16* Kb  = (__bf16*)ws; ws += (size_t)2 * 4 * 2048 * 128 * sizeof(__bf16);  //  4 MB
  __bf16* Vt  = (__bf16*)ws; ws += (size_t)2 * 4 * 2048 * 128 * sizeof(__bf16);  //  4 MB
  __bf16* Yb  = (__bf16*)ws; ws += (size_t)4096 * 2048 * sizeof(__bf16);         // 16 MB

  // 0) one-time operand prep: x -> bf16; W -> bf16 column-major (N x K)
  to_bf16<<<dim3(4096), 256, 0, stream>>>(x, Xb);
  transpose_bf16<<<dim3(3072 / 32, 2048 / 32), 256, 0, stream>>>(W_attn, Wat, 2048, 3072);
  transpose_bf16<<<dim3(2048 / 32, 2048 / 32), 256, 0, stream>>>(W_proj, Wpt, 2048, 2048);

  // 1) qkv = x @ W_attn + b_attn   (async-LDS double-buffered WMMA GEMM)
  gemm_bb<<<dim3(3072 / 128, 4096 / 128), 256, 0, stream>>>(
      Xb, Wat, b_attn, qkv, 4096, 3072, 2048);
  // 2) RoPE(K,V) + bf16 repack (V transposed)
  rope_pack<<<dim3(4096), 256, 0, stream>>>(qkv, fcos, fsin, Qb, Kb, Vt);
  // 3) causal flash attention -> bf16
  attn_fwd<<<dim3(32, 16, 2), 128, 0, stream>>>(Qb, Kb, Vt, Yb);
  // 4) out = y @ W_proj
  gemm_bb<<<dim3(2048 / 128, 4096 / 128), 256, 0, stream>>>(
      Yb, Wpt, nullptr, out, 4096, 2048, 2048);
}
